// InvConv_12120397709625
// MI455X (gfx1250) — matrix operand order repaired and checked by reading, not compiled, not run on hardware
//
#include <hip/hip_runtime.h>
#include <cstddef>

typedef __bf16 bf16_t;
typedef __attribute__((ext_vector_type(16))) __bf16 bf16x16;
typedef __attribute__((ext_vector_type(8)))  float  v8f;

#define NB    32
#define NC    96
#define NMID  512
#define NHW   1024   // 32*32

// ---------- bf16 helpers (round-to-nearest-even) ----------
__device__ __forceinline__ bf16_t f2bf(float f) {
  unsigned u = __builtin_bit_cast(unsigned, f);
  unsigned short h = (unsigned short)((u + 0x7fffu + ((u >> 16) & 1u)) >> 16);
  return __builtin_bit_cast(bf16_t, h);
}
__device__ __forceinline__ float bf2f(bf16_t b) {
  unsigned short h = __builtin_bit_cast(unsigned short, b);
  return __builtin_bit_cast(float, ((unsigned)h) << 16);
}

// ---------- weight packing into WMMA A-fragment layout ----------
// Packed element index = ((mt*NKT + kt)*32 + lane)*16 + i
// A 16x32 fragment layout used here: lane m / m+16 holds row m,
//   i -> kl = 16*half + i (contiguous K order, matches B fragment loads)
// K ordering = tap-major channel-minor: Kg = p*CIN + c, p = ky*KW+kx.
template<int CIN, int COUT, int KHW>
__global__ __launch_bounds__(256) void pack_w(const float* __restrict__ w,
                                              bf16_t* __restrict__ out) {
  constexpr int K = KHW * KHW * CIN;
  constexpr int NKT = K / 32;
  constexpr int NMT = COUT / 16;
  int idx = blockIdx.x * 256 + threadIdx.x;
  if (idx >= NMT * NKT * 512) return;
  int i    = idx & 15;
  int lane = (idx >> 4) & 31;
  int rest = idx >> 9;
  int kt = rest % NKT;
  int mt = rest / NKT;
  int m = lane & 15, half = lane >> 4;
  int v = i >> 1, hb = i & 1;
  int kl = 16 * half + i;
  int kg = kt * 32 + kl;
  int p = kg / CIN, c = kg - p * CIN;
  int o = mt * 16 + m;
  float val = w[((size_t)o * CIN + c) * (KHW * KHW) + p];
  out[idx] = f2bf(val);
}

// ---------- BN batch statistics ----------
__global__ __launch_bounds__(256) void bn_stats_f32(const float* __restrict__ x,
                                                    float* __restrict__ mean,
                                                    float* __restrict__ rstd,
                                                    int C) {
  int c = blockIdx.x, t = threadIdx.x;
  float s = 0.f, ss = 0.f;
  for (int b = 0; b < NB; ++b) {
    const float* p = x + ((size_t)(b * C + c)) * NHW;
    for (int i = t; i < NHW; i += 256) { float v = p[i]; s += v; ss += v * v; }
  }
  __shared__ float sh[256], sh2[256];
  sh[t] = s; sh2[t] = ss; __syncthreads();
  for (int o = 128; o > 0; o >>= 1) {
    if (t < o) { sh[t] += sh[t + o]; sh2[t] += sh2[t + o]; }
    __syncthreads();
  }
  if (t == 0) {
    float n = (float)(NB * NHW);
    float m = sh[0] / n;
    float var = sh2[0] / n - m * m;
    mean[c] = m; rstd[c] = rsqrtf(var + 1e-5f);
  }
}

__global__ __launch_bounds__(256) void bn_stats_bf16(const bf16_t* __restrict__ x,
                                                     float* __restrict__ mean,
                                                     float* __restrict__ rstd,
                                                     int C) {
  int c = blockIdx.x, t = threadIdx.x;
  float s = 0.f, ss = 0.f;
  for (int i = t; i < NB * NHW; i += 256) {
    float v = bf2f(x[(size_t)i * C + c]); s += v; ss += v * v;
  }
  __shared__ float sh[256], sh2[256];
  sh[t] = s; sh2[t] = ss; __syncthreads();
  for (int o = 128; o > 0; o >>= 1) {
    if (t < o) { sh[t] += sh[t + o]; sh2[t] += sh2[t + o]; }
    __syncthreads();
  }
  if (t == 0) {
    float n = (float)(NB * NHW);
    float m = sh[0] / n;
    float var = sh2[0] / n - m * m;
    mean[c] = m; rstd[c] = rsqrtf(var + 1e-5f);
  }
}

// ---------- BN apply on x_cond (f32 NCHW) -> bf16 NHWC ----------
__global__ __launch_bounds__(256) void bn_pack0(const float* __restrict__ x,
                                                const float* __restrict__ mean,
                                                const float* __restrict__ rstd,
                                                const float* __restrict__ g,
                                                const float* __restrict__ be,
                                                bf16_t* __restrict__ out) {
  int idx = blockIdx.x * 256 + threadIdx.x;         // NB*NC*NHW
  if (idx >= NB * NC * NHW) return;
  int pix = idx & (NHW - 1);
  int c   = (idx >> 10) % NC;
  int b   = idx / (NC * NHW);
  float v = x[idx];
  float h = (v - mean[c]) * rstd[c] * g[c] + be[c];
  out[((size_t)(b * NHW + pix)) * NC + c] = f2bf(h);
}

// ---------- BN + ReLU on bf16 NHWC (C = 512) ----------
__global__ __launch_bounds__(256) void bn_apply_relu(const bf16_t* __restrict__ x,
                                                     const float* __restrict__ mean,
                                                     const float* __restrict__ rstd,
                                                     const float* __restrict__ g,
                                                     const float* __restrict__ be,
                                                     bf16_t* __restrict__ out) {
  size_t idx = (size_t)blockIdx.x * 256 + threadIdx.x;  // 32768*512 exact
  int c = (int)(idx & (NMID - 1));
  float v = bf2f(x[idx]);
  float h = (v - mean[c]) * rstd[c] * g[c] + be[c];
  out[idx] = f2bf(h > 0.f ? h : 0.f);
}

// ---------- implicit-GEMM conv: async global->LDS staging + WMMA bf16 ----------
// One block per spatial tile (b, y, 16 columns). The (KHW x (16+2*PAD) x CIN) input
// halo tile is staged into LDS once via gfx1250 async copies (ASYNCcnt), border
// chunks are zero-filled with ds_store. All WAVES waves then run the K loop from
// LDS, each owning MTW 16-row output-channel tiles (WAVES*MTW*16 == COUT).
// MODE 0: relu -> bf16 NHWC   MODE 1: raw -> bf16 NHWC   MODE 2: +bias -> f32 NCHW
template<int CIN, int COUT, int KHW, int MODE, int WAVES, int MTW>
__global__ __launch_bounds__(WAVES * 32) void conv_wmma_lds(
    const bf16_t* __restrict__ act,
    const bf16_t* __restrict__ wfrag,
    const float*  __restrict__ bias,
    bf16_t* __restrict__ outb,
    float*  __restrict__ outf) {
  constexpr int PAD = (KHW == 3) ? 1 : 0;
  constexpr int TW  = 16 + 2 * PAD;      // tile width incl. halo
  constexpr int TH  = KHW;               // tile height
  constexpr int K   = KHW * KHW * CIN;
  constexpr int NKT = K / 32;
  static_assert(WAVES * MTW * 16 == COUT, "tiling must cover COUT");

  __shared__ __align__(16) bf16_t tile[TH * TW * CIN];

  int t = blockIdx.x;                    // 2048 blocks: b*64 + y*2 + xt
  const int xt = t & 1;  t >>= 1;
  const int y  = t & 31;
  const int b  = t >> 5;
  const int xbase = xt * 16;

  const int tid  = threadIdx.x;
  const int wave = tid >> 5;
  const int lane = tid & 31;

  // ---- stage input tile: async DMA for interior, ds_store zeros for padding ----
  constexpr int NCHUNK = TH * TW * CIN / 8;          // 16B chunks
  const unsigned ldsbase = (unsigned)(size_t)(&tile[0]);
  for (int q = tid; q < NCHUNK; q += WAVES * 32) {
    const int e   = q * 8;
    const int cc  = e % CIN;             // CIN multiple of 8: chunk stays in one pixel
    const int cr  = e / CIN;
    const int col = cr % TW;
    const int row = cr / TW;
    const int iy  = y + row - PAD;
    const int ix  = xbase + col - PAD;
    const unsigned lds = ldsbase + (unsigned)e * 2u;
    if ((unsigned)iy < 32u && (unsigned)ix < 32u) {
      const unsigned goff = (unsigned)(((b * NHW + iy * 32 + ix) * CIN + cc) * 2);
      asm volatile("global_load_async_to_lds_b128 %0, %1, %2"
                   :: "v"(lds), "v"(goff), "s"(act) : "memory");
    } else {
      *(uint4*)(&tile[e]) = make_uint4(0u, 0u, 0u, 0u);
    }
  }
  asm volatile("s_wait_asynccnt 0" ::: "memory");
  __syncthreads();

  // ---- K loop: B fragments from LDS, A fragments (packed weights) from L2 ----
  const int n    = lane & 15;
  const int half = lane >> 4;
  const int xcol = xbase + n;

  v8f acc[MTW] = {};
  const bf16_t* wb = wfrag + ((size_t)(wave * MTW) * NKT * 32 + lane) * 16;

  for (int kt = 0; kt < NKT; ++kt) {
    const int kg = kt * 32;
    const int p  = kg / CIN;
    const int cb = kg - p * CIN;
    const int ky = p / KHW;
    const int kx = p - ky * KHW;
    const int col = n + kx;              // 0..TW-1 (halo makes this always valid)

    const bf16x16 bf = *(const bf16x16*)(&tile[(ky * TW + col) * CIN + cb + half * 16]);
#pragma unroll
    for (int m = 0; m < MTW; ++m) {
      bf16x16 af = *(const bf16x16*)(wb + ((size_t)(m * NKT + kt)) * 512);
      acc[m] = __builtin_amdgcn_wmma_f32_16x16x32_bf16(false, af, false, bf,
                                                       (short)0, acc[m], false, false);
    }
    if (kt + 1 < NKT)
      __builtin_prefetch(wb + ((size_t)(kt + 1)) * 512, 0, 1);   // global_prefetch_b8
  }

  // ---- store ----
  if (MODE == 2) {
#pragma unroll
    for (int m = 0; m < MTW; ++m) {
      const int ob = wave * (MTW * 16) + m * 16 + half * 8;
#pragma unroll
      for (int r = 0; r < 8; ++r) {
        int o = ob + r;
        outf[((size_t)(b * COUT + o)) * NHW + y * 32 + xcol] = acc[m][r] + bias[o];
      }
    }
  } else {
    const size_t pixBase = ((size_t)(b * NHW + y * 32 + xcol)) * COUT;
#pragma unroll
    for (int m = 0; m < MTW; ++m) {
      const int ob = wave * (MTW * 16) + m * 16 + half * 8;
      bf16_t tmp[8];
#pragma unroll
      for (int r = 0; r < 8; ++r) {
        float v = acc[m][r];
        if (MODE == 0 && v < 0.f) v = 0.f;
        tmp[r] = f2bf(v);
      }
      *(uint4*)(&outb[pixBase + ob]) = *(const uint4*)tmp;  // 16B aligned (ob % 8 == 0)
    }
  }
}

// ---------- converter: weight[b,c,d] = sum_n wmap[b,c,n]*conv_w[d,n] + conv_b[d] ----------
__global__ __launch_bounds__(128) void converter_k(const float* __restrict__ wmap,
                                                   const float* __restrict__ cw,
                                                   const float* __restrict__ cb,
                                                   float* __restrict__ weight) {
  int b = blockIdx.x / NC, c = blockIdx.x % NC;
  __shared__ float row[NHW];
  const float* src = wmap + ((size_t)(b * NC + c)) * NHW;
  for (int i = threadIdx.x; i < NHW; i += 128) row[i] = src[i];
  __syncthreads();
  int d = threadIdx.x;
  if (d < NC) {
    const float* w = cw + (size_t)d * NHW;
    float acc = 0.f;
    for (int i = 0; i < NHW; ++i) acc = fmaf(row[i], w[i], acc);
    weight[((size_t)(b * NC) + c) * NC + d] = acc + cb[d];
  }
}

// ---------- per-sample logabsdet via LU with partial pivoting (in LDS) ----------
__global__ __launch_bounds__(96) void slogdet_k(const float* __restrict__ weight,
                                                const float* __restrict__ sldj_in,
                                                float* __restrict__ sldj_out) {
  const int b = blockIdx.x;
  const int t = threadIdx.x;                // 96 threads, thread t owns column t
  __shared__ float A[NC][NC + 1];
  __shared__ float pv[NC];
  __shared__ int   piv;
  __shared__ float lacc;
  const float* src = weight + (size_t)b * NC * NC;
  for (int i = 0; i < NC; ++i) A[i][t] = src[i * NC + t];
  if (t == 0) lacc = 0.f;
  __syncthreads();
  for (int k = 0; k < NC; ++k) {
    pv[t] = (t >= k) ? fabsf(A[t][k]) : -1.f;
    __syncthreads();
    if (t == 0) {
      int best = k; float bv = pv[k];
      for (int i = k + 1; i < NC; ++i) if (pv[i] > bv) { bv = pv[i]; best = i; }
      piv = best;
    }
    __syncthreads();
    int p = piv;
    if (p != k) { float tmp = A[k][t]; A[k][t] = A[p][t]; A[p][t] = tmp; }
    __syncthreads();
    float dkk = A[k][k];
    if (t == 0) lacc += logf(fabsf(dkk));
    if (t > k) A[t][k] = A[t][k] / dkk;
    __syncthreads();
    if (t > k) {
      float akt = A[k][t];
      for (int i = k + 1; i < NC; ++i) A[i][t] -= A[i][k] * akt;
    }
    __syncthreads();
  }
  if (t == 0) sldj_out[b] = sldj_in[b] + lacc * (float)NHW;
}

// ---------- z[b,o,p] = sum_j weight[b,o,j] * x[b,j,p] (f32, memory-bound) ----------
__global__ __launch_bounds__(256) void apply_z(const float* __restrict__ weight,
                                               const float* __restrict__ x,
                                               float* __restrict__ z) {
  int o = blockIdx.x % NC, b = blockIdx.x / NC;
  __shared__ float wrow[NC];
  if (threadIdx.x < NC) wrow[threadIdx.x] = weight[((size_t)b * NC + o) * NC + threadIdx.x];
  __syncthreads();
  for (int p = threadIdx.x; p < NHW; p += 256) {
    const float* xb = x + (size_t)b * NC * NHW + p;
    float acc = 0.f;
#pragma unroll 4
    for (int j = 0; j < NC; ++j) acc = fmaf(wrow[j], xb[(size_t)j * NHW], acc);
    z[((size_t)(b * NC) + o) * NHW + p] = acc;
  }
}

extern "C" void kernel_launch(void* const* d_in, const int* in_sizes, int n_in,
                              void* d_out, int out_size, void* d_ws, size_t ws_size,
                              hipStream_t stream) {
  (void)in_sizes; (void)n_in; (void)out_size; (void)ws_size;
  const float* x      = (const float*)d_in[0];
  const float* x_cond = (const float*)d_in[1];
  const float* sldj   = (const float*)d_in[2];
  const float* g1     = (const float*)d_in[3];
  const float* b1     = (const float*)d_in[4];
  const float* w_in   = (const float*)d_in[5];
  const float* w_mid1 = (const float*)d_in[6];
  const float* g2     = (const float*)d_in[7];
  const float* b2     = (const float*)d_in[8];
  const float* w_mid2 = (const float*)d_in[9];
  const float* g3     = (const float*)d_in[10];
  const float* b3     = (const float*)d_in[11];
  const float* w_out  = (const float*)d_in[12];
  const float* b_out  = (const float*)d_in[13];
  const float* conv_w = (const float*)d_in[14];
  const float* conv_b = (const float*)d_in[15];

  char* ws = (char*)d_ws;
  float* stats = (float*)ws;
  float* m0 = stats +    0, *r0 = stats +  512;
  float* m2 = stats + 1024, *r2 = stats + 1536;
  float* m3 = stats + 2048, *r3 = stats + 2560;

  size_t off = 16384;
  bf16_t* pw_in = (bf16_t*)(ws + off); off += (size_t)512 * 864  * 2; off = (off + 255) & ~(size_t)255;
  bf16_t* pw_m1 = (bf16_t*)(ws + off); off += (size_t)512 * 4608 * 2; off = (off + 255) & ~(size_t)255;
  bf16_t* pw_m2 = (bf16_t*)(ws + off); off += (size_t)512 * 512  * 2; off = (off + 255) & ~(size_t)255;
  bf16_t* pw_o  = (bf16_t*)(ws + off); off += (size_t)96  * 4608 * 2; off = (off + 255) & ~(size_t)255;
  const size_t BUF = (size_t)NB * NHW * NMID * 2;   // 32 MiB
  bf16_t* buf0 = (bf16_t*)(ws + off); off += BUF;
  bf16_t* buf1 = (bf16_t*)(ws + off); off += BUF;
  bf16_t* buf2 = (bf16_t*)(ws + off); off += BUF;
  bf16_t* buf3 = (bf16_t*)(ws + off); off += BUF;
  float*  wmat = (float*)(ws + off);  off += (size_t)NB * NC * NC * 4;

  // buffer reuse schedule (in-order stream => safe):
  bf16_t* act0 = buf0;            // [32768, 96]   bf16 NHWC
  bf16_t* act1 = buf1;            // [32768, 512]
  bf16_t* h2   = buf2;            // conv_mid1 raw
  bf16_t* act2 = buf3;
  bf16_t* h3   = buf0;            // reuse (act0 dead)
  bf16_t* act3 = buf1;            // reuse (act1 dead)
  float*  wmap = (float*)buf2;    // reuse (h2 dead), 12.6 MB f32 NCHW

  float* zout     = (float*)d_out;                 // [32,96,1024]
  float* sldj_out = zout + (size_t)NB * NC * NHW;  // [32]

  // 1) pack conv weights into WMMA A-fragment layout
  pack_w<96, 512, 3><<<(32 * 27 * 512) / 256, 256, 0, stream>>>(w_in,   pw_in);
  pack_w<512, 512, 3><<<(32 * 144 * 512) / 256, 256, 0, stream>>>(w_mid1, pw_m1);
  pack_w<512, 512, 1><<<(32 * 16 * 512) / 256, 256, 0, stream>>>(w_mid2, pw_m2);
  pack_w<512, 96, 3><<<(6 * 144 * 512) / 256, 256, 0, stream>>>(w_out,  pw_o);

  // 2) BN(x_cond) -> bf16 NHWC
  bn_stats_f32<<<NC, 256, 0, stream>>>(x_cond, m0, r0, NC);
  bn_pack0<<<(NB * NC * NHW) / 256, 256, 0, stream>>>(x_cond, m0, r0, g1, b1, act0);

  const int NBLK = NB * 32 * 2;   // one block per (b, y, 16-col tile)

  // 3) conv_in 3x3 (96->512) + ReLU            [async->LDS + WMMA]
  conv_wmma_lds<96, 512, 3, 0, 4, 8><<<NBLK, 128, 0, stream>>>(act0, pw_in, nullptr, act1, nullptr);

  // 4) conv_mid1 3x3 (512->512)                [async->LDS + WMMA]
  conv_wmma_lds<512, 512, 3, 1, 4, 8><<<NBLK, 128, 0, stream>>>(act1, pw_m1, nullptr, h2, nullptr);
  bn_stats_bf16<<<NMID, 256, 0, stream>>>(h2, m2, r2, NMID);
  bn_apply_relu<<<(NB * NHW * NMID) / 256, 256, 0, stream>>>(h2, m2, r2, g2, b2, act2);

  // 5) conv_mid2 1x1 (512->512)                [async->LDS + WMMA]
  conv_wmma_lds<512, 512, 1, 1, 4, 8><<<NBLK, 128, 0, stream>>>(act2, pw_m2, nullptr, h3, nullptr);
  bn_stats_bf16<<<NMID, 256, 0, stream>>>(h3, m3, r3, NMID);
  bn_apply_relu<<<(NB * NHW * NMID) / 256, 256, 0, stream>>>(h3, m3, r3, g3, b3, act3);

  // 6) conv_out 3x3 (512->96) + bias -> wmap f32 NCHW   [async->LDS + WMMA]
  conv_wmma_lds<512, 96, 3, 2, 3, 2><<<NBLK, 96, 0, stream>>>(act3, pw_o, b_out, nullptr, wmap);

  // 7) converter -> per-sample 96x96 weight (f32 for slogdet stability)
  converter_k<<<NB * NC, 128, 0, stream>>>(wmap, conv_w, conv_b, wmat);

  // 8) slogdet + sldj
  slogdet_k<<<NB, 96, 0, stream>>>(wmat, sldj, sldj_out);

  // 9) z = weight x  (per-sample channel mix, f32, memory-bound)
  apply_z<<<NB * NC, 256, 0, stream>>>(wmat, x, zout);
}
